// StrongParT_4561255268373
// MI455X (gfx1250) — matrix operand order repaired and checked by reading, not compiled, not run on hardware
//
#include <hip/hip_runtime.h>
#include <math.h>

typedef __attribute__((ext_vector_type(16))) _Float16 v16h;
typedef __attribute__((ext_vector_type(8)))  float    v8f;
typedef _Float16 h16;

#define E_   256
#define H_   8
#define L_   6
#define DFF_ 1024
#define B_   16
#define N_   512
#define BN_  8192

union F16x16 { v16h v; h16 e[16]; float4 q[2]; };

__device__ __forceinline__ float gelu_f(float x) {
  return 0.5f * x * (1.f + erff(x * 0.70710678118654752f));
}

__device__ __forceinline__ v8f wmma_f16(v16h a, v16h b, v8f c) {
  return __builtin_amdgcn_wmma_f32_16x16x32_f16(false, a, false, b, (short)0, c, false, false);
}

// A fragment: 16(M) x 32(K) f16, source row-major with leading dim ld.
// Order used here (not the ISA table): lane m=lane&15, g=lane>>4; halfs 0..15 -> K = g*16+0..15 (contiguous).
__device__ __forceinline__ v16h load_afrag(const h16* base, int ld, int lane) {
  int m = lane & 15, g = lane >> 4;
  const h16* p = base + m * ld + g * 16;
  F16x16 u;
  u.q[0] = *(const float4*)(p);
  u.q[1] = *(const float4*)(p + 8);
  return u.v;
}

// Packed B tile: 512 halfs, lane-major (lane*16 + h), pre-packed by pack_b_k.
__device__ __forceinline__ v16h load_b16(const h16* tile, int lane) {
  F16x16 u;
  const h16* p = tile + lane * 16;
  u.q[0] = *(const float4*)(p);
  u.q[1] = *(const float4*)(p + 8);
  return u.v;
}

// B fragment where, for this lane, halfs 0..15 are contiguous in memory.
__device__ __forceinline__ v16h load_contig16(const h16* p) {
  F16x16 u;
  u.q[0] = *(const float4*)(p);
  u.q[1] = *(const float4*)(p + 8);
  return u.v;
}

// B fragment 32(K) x 16(N): element (k,n) = base[k*strideK + n]; lane n=lane&15, k=h+16*(lane>>4)
__device__ __forceinline__ v16h load_bgather(const h16* base, int strideK, int lane) {
  int n = lane & 15, g = lane >> 4;
  F16x16 u;
#pragma unroll
  for (int h = 0; h < 16; ++h) u.e[h] = base[(h + 16 * g) * strideK + n];
  return u.v;
}

// ---------------- weight packing: f32 (K x Nsrc) -> f16 fragment tiles (Kp x Np) -------------
__global__ void pack_b_k(const float* __restrict__ src, h16* __restrict__ dst,
                         int K, int Kp, int Nsrc, int Np) {
  int lane = threadIdx.x;
  int nt = blockIdx.x, kt = blockIdx.y;
  int n = nt * 16 + (lane & 15);
  int g = lane >> 4;
  h16* out = dst + ((size_t)nt * (Kp / 32) + kt) * 512 + lane * 16;
#pragma unroll
  for (int h = 0; h < 16; ++h) {
    int k = kt * 32 + h + 16 * g;
    float v = (k < K && n < Nsrc) ? src[(size_t)k * Nsrc + n] : 0.f;
    out[h] = (h16)v;
  }
}

// ---------------- generic WMMA GEMM: C = act(A(MxK,f16) @ Bpacked(KxN,f16) + bias) -----------
__global__ void gemm_k(const h16* __restrict__ A, const h16* __restrict__ Bp,
                       const float* __restrict__ bias,
                       float* __restrict__ Cf, h16* __restrict__ Ch,
                       int M, int N, int K, int act) {
  int lane = threadIdx.x;
  int nt = blockIdx.x, mt = blockIdx.y;
  int ks = K >> 5;
  const h16* Abase = A + (size_t)(mt * 16) * K;
  const h16* Btile = Bp + (size_t)nt * ks * 512;
  v8f c = {};
  for (int s = 0; s < ks; ++s) {
    v16h a = load_afrag(Abase + s * 32, K, lane);
    v16h b = load_b16(Btile + s * 512, lane);
    c = wmma_f16(a, b, c);
  }
  int n = lane & 15, g = lane >> 4;
  int col = nt * 16 + n;
  float bv = bias ? bias[col] : 0.f;
#pragma unroll
  for (int r = 0; r < 8; ++r) {
    int m = mt * 16 + r + 8 * g;
    float v = c[r] + bv;
    if (act == 1) v = gelu_f(v);
    else if (act == 2) v = fmaxf(v, 0.f);
    if (Cf) Cf[(size_t)m * N + col] = v;
    if (Ch) Ch[(size_t)m * N + col] = (h16)v;
  }
}

// ---------------- pair-feature normalization (mean |.| / mean dist over NxN per batch) -------
__global__ void zero_k(float* p, int n) {
  int i = blockIdx.x * 256 + threadIdx.x;
  if (i < n) p[i] = 0.f;
}

__global__ void pairsum_k(const float* __restrict__ hits, float* __restrict__ sums) {
  int b = blockIdx.y;
  int base = blockIdx.x * 4096 + threadIdx.x;
  float acc[6] = {0, 0, 0, 0, 0, 0};
  for (int i = 0; i < 16; ++i) {
    int p = base + i * 256;
    int q = p >> 9, k = p & 511;
    const float* hq = hits + ((size_t)b * N_ + q) * 5;
    const float* hk = hits + ((size_t)b * N_ + k) * 5;
    float d0 = hq[0] - hk[0], d1 = hq[1] - hk[1], d2 = hq[2] - hk[2];
    float d3 = hq[3] - hk[3], d4 = hq[4] - hk[4];
    acc[0] += fabsf(d0); acc[1] += fabsf(d1);
    acc[2] += sqrtf(d0 * d0 + d1 * d1 + 1e-6f);
    acc[3] += fabsf(d2); acc[4] += fabsf(d3); acc[5] += fabsf(d4);
  }
  __shared__ float red[256];
  int t = threadIdx.x;
  for (int j = 0; j < 6; ++j) {
    red[t] = acc[j]; __syncthreads();
    for (int s = 128; s > 0; s >>= 1) { if (t < s) red[t] += red[t + s]; __syncthreads(); }
    if (t == 0) atomicAdd(&sums[b * 6 + j], red[0]);
    __syncthreads();
  }
}

__global__ void norm_fin_k(const float* sums, float* norm) {
  int i = threadIdx.x;
  if (i < B_ * 6) norm[i] = sums[i] / (512.f * 512.f) + 1e-6f;
}

// ---------------- embed: feat = gelu(LN(hits@Ew+be)) + gelu(pos@Pw+bp) -----------------------
__global__ void embed_k(const float* __restrict__ hits,
                        const float* __restrict__ ew, const float* __restrict__ eb,
                        const float* __restrict__ lns, const float* __restrict__ lnb,
                        const float* __restrict__ pw, const float* __restrict__ pb,
                        float* __restrict__ featf, h16* __restrict__ feath) {
  int row = blockIdx.x, e = threadIdx.x;
  const float* hp = hits + (size_t)row * 5;
  float x = eb[e];
#pragma unroll
  for (int i = 0; i < 5; ++i) x += hp[i] * ew[i * E_ + e];
  __shared__ float red[256];
  red[e] = x; __syncthreads();
  for (int s = 128; s > 0; s >>= 1) { if (e < s) red[e] += red[e + s]; __syncthreads(); }
  float mean = red[0] * (1.f / E_); __syncthreads();
  float d = x - mean;
  red[e] = d * d; __syncthreads();
  for (int s = 128; s > 0; s >>= 1) { if (e < s) red[e] += red[e + s]; __syncthreads(); }
  float var = red[0] * (1.f / E_);
  float y = d * rsqrtf(var + 1e-5f) * lns[e] + lnb[e];
  float p = pb[e] + hp[0] * pw[e] + hp[1] * pw[E_ + e];
  float out = gelu_f(y) + gelu_f(p);
  featf[(size_t)row * E_ + e] = out;
  feath[(size_t)row * E_ + e] = (h16)out;
}

// ---------------- fused pair-bias MLP: bias[b,q,k,h] = relu(pf@W1+b1)@W2+b2 ------------------
// grid (N/16, N/4, B), block 128 (4 waves); wave w handles q = blockIdx.y*4+w, 16 k-values.
__global__ void pairbias_k(const float* __restrict__ hits, const float* __restrict__ norm,
                           const h16* __restrict__ pw1p, const h16* __restrict__ pw2p,
                           const float* __restrict__ pb1, const float* __restrict__ pb2,
                           h16* __restrict__ bias_out) {
  int wv = threadIdx.x >> 5, lane = threadIdx.x & 31;
  int b = blockIdx.z;
  int q = blockIdx.y * 4 + wv;
  int k0 = blockIdx.x * 16;
  __shared__ __align__(16) h16 hidsh[4][16 * 264];
  h16* hid = hidsh[wv];

  F16x16 ua;
#pragma unroll
  for (int h = 0; h < 16; ++h) ua.e[h] = (h16)0.f;
  if (lane < 16) {
    const float* hq = hits + ((size_t)b * N_ + q) * 5;
    const float* hk = hits + ((size_t)b * N_ + k0 + lane) * 5;
    float d0 = hq[0] - hk[0], d1 = hq[1] - hk[1], d2 = hq[2] - hk[2];
    float d3 = hq[3] - hk[3], d4 = hq[4] - hk[4];
    float dist = sqrtf(d0 * d0 + d1 * d1 + 1e-6f);
    const float* nb = norm + b * 6;
    ua.e[0] = (h16)(d0 / nb[0]);
    ua.e[1] = (h16)(d1 / nb[1]);
    ua.e[2] = (h16)(dist / nb[2]);
    ua.e[3] = (h16)(d2 / nb[3]);
    ua.e[4] = (h16)(d3 / nb[4]);
    ua.e[5] = (h16)(d4 / nb[5]);
  }
  v16h a = ua.v;
  int n = lane & 15, g = lane >> 4;

  // stage 1: hidden = relu(pf @ W1 + b1), 16x256, kept in LDS as f16
  for (int nt = 0; nt < 16; ++nt) {
    v16h bfr = load_b16(pw1p + nt * 512, lane);
    v8f c = {};
    c = wmma_f16(a, bfr, c);
    int col = nt * 16 + n;
    float b1v = pb1[col];
#pragma unroll
    for (int r = 0; r < 8; ++r) {
      float v = fmaxf(c[r] + b1v, 0.f);
      hid[(r + 8 * g) * 264 + col] = (h16)v;
    }
  }
  __syncthreads();

  // stage 2: bias = hidden @ W2 + b2 (N padded 8->16)
  v8f c2 = {};
  for (int s = 0; s < 8; ++s) {
    v16h a2 = load_afrag(hid + s * 32, 264, lane);
    v16h b2 = load_b16(pw2p + s * 512, lane);
    c2 = wmma_f16(a2, b2, c2);
  }
  if (n < 8) {
    float b2v = pb2[n];
#pragma unroll
    for (int r = 0; r < 8; ++r) {
      int kk = k0 + r + 8 * g;
      bias_out[(((size_t)b * N_ + q) * N_ + kk) * H_ + n] = (h16)(c2[r] + b2v);
    }
  }
}

// ---------------- attention: per (b, head, 16-query tile), full 512-key softmax --------------
__global__ void attn_k(const h16* __restrict__ qkvh, const h16* __restrict__ biash,
                       const unsigned char* __restrict__ mask, h16* __restrict__ attnh) {
  int lane = threadIdx.x;
  int qt = blockIdx.x, hh = blockIdx.y, b = blockIdx.z;
  int q0 = qt * 16;
  __shared__ __align__(16) float lg[16 * 516];
  __shared__ __align__(16) h16 pr[16 * 520];
  __shared__ float rowsum[16];
  int n = lane & 15, g = lane >> 4;
  const float scale = 0.1767766952966369f; // 1/sqrt(32)

  const h16* Qbase = qkvh + ((size_t)b * N_ + q0) * 768 + hh * 32;
  v16h aq = load_afrag(Qbase, 768, lane);

  // logits: Q(16x32) @ K^T(32x16) per key tile
  for (int kt = 0; kt < 32; ++kt) {
    int key = kt * 16 + n;
    const h16* Kb = qkvh + ((size_t)b * N_ + key) * 768 + 256 + hh * 32 + g * 16;
    v16h bf = load_contig16(Kb);
    v8f c = {};
    c = wmma_f16(aq, bf, c);
    bool mk = mask[b * N_ + key] != 0;
#pragma unroll
    for (int r = 0; r < 8; ++r) {
      int m = r + 8 * g;
      float v;
      if (mk) {
        float bv = (float)biash[(((size_t)b * N_ + q0 + m) * N_ + key) * H_ + hh];
        v = c[r] * scale + bv;
      } else {
        v = -1e9f;
      }
      lg[m * 516 + key] = v;
    }
  }
  __syncthreads();

  // stable softmax per row; store unnormalized exp as f16, fold 1/sum into epilogue
  if (lane < 16) {
    float mx = -3.4e38f;
    for (int k = 0; k < N_; ++k) mx = fmaxf(mx, lg[lane * 516 + k]);
    float s = 0.f;
    for (int k = 0; k < N_; ++k) {
      float p = __expf(lg[lane * 516 + k] - mx);
      s += p;
      pr[lane * 520 + k] = (h16)p;
    }
    rowsum[lane] = s;
  }
  __syncthreads();

  // out = probs(16x512) @ V(512x32)
  for (int nt = 0; nt < 2; ++nt) {
    v8f c = {};
    for (int s = 0; s < 16; ++s) {
      v16h ap = load_afrag(pr + s * 32, 520, lane);
      const h16* Vb = qkvh + ((size_t)b * N_ + s * 32) * 768 + 512 + hh * 32 + nt * 16;
      v16h bf = load_bgather(Vb, 768, lane);
      c = wmma_f16(ap, bf, c);
    }
#pragma unroll
    for (int r = 0; r < 8; ++r) {
      int m = r + 8 * g;
      float inv = 1.f / rowsum[m];
      attnh[((size_t)b * N_ + q0 + m) * E_ + hh * 32 + nt * 16 + n] = (h16)(c[r] * inv);
    }
  }
}

// ---------------- residual + LayerNorm ------------------------------------------------------
__global__ void ln_k(const float* __restrict__ a, const float* __restrict__ res,
                     const float* __restrict__ s, const float* __restrict__ bb,
                     float* __restrict__ outf, h16* __restrict__ outh) {
  int row = blockIdx.x, t = threadIdx.x;
  __shared__ float red[256];
  size_t idx = (size_t)row * E_ + t;
  float x = a[idx] + (res ? res[idx] : 0.f);
  red[t] = x; __syncthreads();
  for (int st = 128; st > 0; st >>= 1) { if (t < st) red[t] += red[t + st]; __syncthreads(); }
  float mean = red[0] * (1.f / E_); __syncthreads();
  float d = x - mean;
  red[t] = d * d; __syncthreads();
  for (int st = 128; st > 0; st >>= 1) { if (t < st) red[t] += red[t + st]; __syncthreads(); }
  float var = red[0] * (1.f / E_);
  float y = d * rsqrtf(var + 1e-5f) * s[t] + bb[t];
  if (outf) outf[idx] = y;
  if (outh) outh[idx] = (h16)y;
}

// ---------------- masked mean/max pooling over N --------------------------------------------
__global__ void pool_k(const float* __restrict__ feat, const unsigned char* __restrict__ mask,
                       float* __restrict__ meanf, float* __restrict__ maxf) {
  int b = blockIdx.x, e = threadIdx.x;
  float s = 0.f, mx = -1e9f, cnt = 0.f;
  for (int nn = 0; nn < N_; ++nn) {
    float v = feat[((size_t)b * N_ + nn) * E_ + e];
    if (mask[b * N_ + nn] != 0) { s += v; mx = fmaxf(mx, v); cnt += 1.f; }
  }
  float mean = s / (cnt + 1e-6f);
  if (cnt == 0.f) mx = mean;
  meanf[b * E_ + e] = mean;
  maxf[b * E_ + e] = mx;
}

// ---------------- param head: gelu -> LN -> gelu --------------------------------------------
__global__ void pf_k(const float* __restrict__ params,
                     const float* __restrict__ w1, const float* __restrict__ b1,
                     const float* __restrict__ lns, const float* __restrict__ lnb,
                     const float* __restrict__ w2, const float* __restrict__ b2,
                     float* __restrict__ pf) {
  int b = blockIdx.x, e = threadIdx.x;
  const float* pp = params + b * 5;
  float x = b1[e];
#pragma unroll
  for (int i = 0; i < 5; ++i) x += pp[i] * w1[i * E_ + e];
  x = gelu_f(x);
  __shared__ float red[256];
  __shared__ float lnv[256];
  red[e] = x; __syncthreads();
  for (int st = 128; st > 0; st >>= 1) { if (e < st) red[e] += red[e + st]; __syncthreads(); }
  float mean = red[0] * (1.f / E_); __syncthreads();
  float d = x - mean;
  red[e] = d * d; __syncthreads();
  for (int st = 128; st > 0; st >>= 1) { if (e < st) red[e] += red[e + st]; __syncthreads(); }
  float var = red[0] * (1.f / E_);
  float y = d * rsqrtf(var + 1e-5f) * lns[e] + lnb[e];
  lnv[e] = y; __syncthreads();
  float z = b2[e];
  for (int i = 0; i < E_; ++i) z += lnv[i] * w2[i * E_ + e];
  pf[b * E_ + e] = gelu_f(z);
}

// ---------------- classifier ----------------------------------------------------------------
__global__ void cls1_k(const float* __restrict__ maxf, const float* __restrict__ meanf,
                       const float* __restrict__ pf, const float* __restrict__ w,
                       const float* __restrict__ bb, float* __restrict__ out) {
  int b = blockIdx.x, j = threadIdx.x; // 512 threads
  float s = bb[j];
  for (int i = 0; i < 256; ++i) s += maxf[b * 256 + i] * w[(size_t)i * 512 + j];
  for (int i = 0; i < 256; ++i) s += meanf[b * 256 + i] * w[(size_t)(256 + i) * 512 + j];
  for (int i = 0; i < 256; ++i) s += pf[b * 256 + i] * w[(size_t)(512 + i) * 512 + j];
  out[b * 512 + j] = gelu_f(s);
}

__global__ void cls2_k(const float* __restrict__ in, const float* __restrict__ w,
                       const float* __restrict__ bb, float* __restrict__ out) {
  int b = blockIdx.x, j = threadIdx.x; // 256 threads
  float s = bb[j];
  for (int i = 0; i < 512; ++i) s += in[b * 512 + i] * w[(size_t)i * 256 + j];
  out[b * 256 + j] = gelu_f(s);
}

__global__ void cls3_k(const float* __restrict__ in, const float* __restrict__ w,
                       const float* __restrict__ bb, float* __restrict__ out) {
  int b = blockIdx.x, j = threadIdx.x; // 32 threads, 2 active
  if (j < 2) {
    float s = bb[j];
    for (int i = 0; i < 256; ++i) s += in[b * 256 + i] * w[i * 2 + j];
    out[b * 2 + j] = s;
  }
}

// =============================================================================================
extern "C" void kernel_launch(void* const* d_in, const int* in_sizes, int n_in,
                              void* d_out, int out_size, void* d_ws, size_t ws_size,
                              hipStream_t stream) {
  (void)in_sizes; (void)n_in; (void)out_size; (void)ws_size;
  const float* hits       = (const float*)d_in[0];
  const unsigned char* mk = (const unsigned char*)d_in[1]; // jax bool -> 1 byte
  const float* params     = (const float*)d_in[2];
  const float* embed_w    = (const float*)d_in[3];
  const float* embed_b    = (const float*)d_in[4];
  const float* embed_ln_s = (const float*)d_in[5];
  const float* embed_ln_b = (const float*)d_in[6];
  const float* pos_w      = (const float*)d_in[7];
  const float* pos_b      = (const float*)d_in[8];
  const float* qkv_w      = (const float*)d_in[9];
  const float* out_w      = (const float*)d_in[10];
  const float* out_b      = (const float*)d_in[11];
  const float* pair_w1    = (const float*)d_in[12];
  const float* pair_b1    = (const float*)d_in[13];
  const float* pair_w2    = (const float*)d_in[14];
  const float* pair_b2    = (const float*)d_in[15];
  const float* ffn_w1     = (const float*)d_in[16];
  const float* ffn_b1     = (const float*)d_in[17];
  const float* ffn_w2     = (const float*)d_in[18];
  const float* ffn_b2     = (const float*)d_in[19];
  const float* n1_s       = (const float*)d_in[20];
  const float* n1_b       = (const float*)d_in[21];
  const float* n2_s       = (const float*)d_in[22];
  const float* n2_b       = (const float*)d_in[23];
  const float* pfc_w1     = (const float*)d_in[24];
  const float* pfc_b1     = (const float*)d_in[25];
  const float* pfc_ln_s   = (const float*)d_in[26];
  const float* pfc_ln_b   = (const float*)d_in[27];
  const float* pfc_w2     = (const float*)d_in[28];
  const float* pfc_b2     = (const float*)d_in[29];
  const float* cls_w1     = (const float*)d_in[30];
  const float* cls_b1     = (const float*)d_in[31];
  const float* cls_w2     = (const float*)d_in[32];
  const float* cls_b2     = (const float*)d_in[33];
  const float* cls_w3     = (const float*)d_in[34];
  const float* cls_b3     = (const float*)d_in[35];

  // ---- workspace layout (bump allocator, 256B aligned) ----
  char* ws = (char*)d_ws;
  size_t off = 0;
  auto take = [&](size_t bytes) -> void* {
    void* p = ws + off;
    off = (off + bytes + 255) & ~(size_t)255;
    return p;
  };
  h16* h_qkv  = (h16*)take((size_t)L_ * E_ * 3 * E_ * 2);
  h16* h_out  = (h16*)take((size_t)L_ * E_ * E_ * 2);
  h16* h_ffn1 = (h16*)take((size_t)L_ * E_ * DFF_ * 2);
  h16* h_ffn2 = (h16*)take((size_t)L_ * DFF_ * E_ * 2);
  h16* h_pw1  = (h16*)take((size_t)L_ * 32 * E_ * 2);
  h16* h_pw2  = (h16*)take((size_t)L_ * E_ * 16 * 2);
  float* feat_f = (float*)take((size_t)BN_ * E_ * 4);
  h16*   feat_h = (h16*)take((size_t)BN_ * E_ * 2);
  h16*   qkv_h  = (h16*)take((size_t)BN_ * 3 * E_ * 2);
  h16*   bias_h = (h16*)take((size_t)B_ * N_ * N_ * H_ * 2);
  h16*   attn_h = (h16*)take((size_t)BN_ * E_ * 2);
  float* proj_f = (float*)take((size_t)BN_ * E_ * 4);
  float* x_f    = (float*)take((size_t)BN_ * E_ * 4);
  h16*   x_h    = (h16*)take((size_t)BN_ * E_ * 2);
  h16*   hid_h  = (h16*)take((size_t)BN_ * DFF_ * 2);
  float* ffn_f  = (float*)take((size_t)BN_ * E_ * 4);
  float* psum   = (float*)take(B_ * 6 * 4);
  float* pnorm  = (float*)take(B_ * 6 * 4);
  float* meanf  = (float*)take(B_ * E_ * 4);
  float* maxf   = (float*)take(B_ * E_ * 4);
  float* pfb    = (float*)take(B_ * E_ * 4);
  float* c1     = (float*)take(B_ * 512 * 4);
  float* c2     = (float*)take(B_ * 256 * 4);

  // ---- pack weights to f16 WMMA fragment tiles ----
  for (int l = 0; l < L_; ++l) {
    pack_b_k<<<dim3(48, 8), 32, 0, stream>>>(qkv_w + (size_t)l * E_ * 3 * E_,
                                             h_qkv + (size_t)l * E_ * 3 * E_, E_, E_, 3 * E_, 3 * E_);
    pack_b_k<<<dim3(16, 8), 32, 0, stream>>>(out_w + (size_t)l * E_ * E_,
                                             h_out + (size_t)l * E_ * E_, E_, E_, E_, E_);
    pack_b_k<<<dim3(64, 8), 32, 0, stream>>>(ffn_w1 + (size_t)l * E_ * DFF_,
                                             h_ffn1 + (size_t)l * E_ * DFF_, E_, E_, DFF_, DFF_);
    pack_b_k<<<dim3(16, 32), 32, 0, stream>>>(ffn_w2 + (size_t)l * DFF_ * E_,
                                              h_ffn2 + (size_t)l * DFF_ * E_, DFF_, DFF_, E_, E_);
    pack_b_k<<<dim3(16, 1), 32, 0, stream>>>(pair_w1 + (size_t)l * 6 * E_,
                                             h_pw1 + (size_t)l * 32 * E_, 6, 32, E_, E_);
    pack_b_k<<<dim3(1, 8), 32, 0, stream>>>(pair_w2 + (size_t)l * E_ * H_,
                                            h_pw2 + (size_t)l * E_ * 16, E_, E_, H_, 16);
  }

  // ---- pair-feature normalization constants ----
  zero_k<<<1, 256, 0, stream>>>(psum, B_ * 6);
  pairsum_k<<<dim3(64, B_), 256, 0, stream>>>(hits, psum);
  norm_fin_k<<<1, 128, 0, stream>>>(psum, pnorm);

  // ---- embedding ----
  embed_k<<<BN_, 256, 0, stream>>>(hits, embed_w, embed_b, embed_ln_s, embed_ln_b,
                                   pos_w, pos_b, feat_f, feat_h);

  // ---- transformer layers ----
  for (int l = 0; l < L_; ++l) {
    pairbias_k<<<dim3(N_ / 16, N_ / 4, B_), 128, 0, stream>>>(
        hits, pnorm, h_pw1 + (size_t)l * 32 * E_, h_pw2 + (size_t)l * E_ * 16,
        pair_b1 + (size_t)l * E_, pair_b2 + (size_t)l * H_, bias_h);
    gemm_k<<<dim3(48, 512), 32, 0, stream>>>(feat_h, h_qkv + (size_t)l * E_ * 3 * E_,
                                             nullptr, nullptr, qkv_h, BN_, 3 * E_, E_, 0);
    attn_k<<<dim3(N_ / 16, H_, B_), 32, 0, stream>>>(qkv_h, bias_h, mk, attn_h);
    gemm_k<<<dim3(16, 512), 32, 0, stream>>>(attn_h, h_out + (size_t)l * E_ * E_,
                                             out_b + (size_t)l * E_, proj_f, nullptr, BN_, E_, E_, 0);
    ln_k<<<BN_, 256, 0, stream>>>(feat_f, proj_f, n1_s + (size_t)l * E_, n1_b + (size_t)l * E_,
                                  x_f, x_h);
    gemm_k<<<dim3(64, 512), 32, 0, stream>>>(x_h, h_ffn1 + (size_t)l * E_ * DFF_,
                                             ffn_b1 + (size_t)l * DFF_, nullptr, hid_h,
                                             BN_, DFF_, E_, 1);
    gemm_k<<<dim3(16, 512), 32, 0, stream>>>(hid_h, h_ffn2 + (size_t)l * DFF_ * E_,
                                             ffn_b2 + (size_t)l * E_, ffn_f, nullptr,
                                             BN_, E_, DFF_, 0);
    ln_k<<<BN_, 256, 0, stream>>>(x_f, ffn_f, n2_s + (size_t)l * E_, n2_b + (size_t)l * E_,
                                  feat_f, feat_h);
  }

  // ---- pooling + heads ----
  pool_k<<<B_, 256, 0, stream>>>(feat_f, mk, meanf, maxf);
  pf_k<<<B_, 256, 0, stream>>>(params, pfc_w1, pfc_b1, pfc_ln_s, pfc_ln_b, pfc_w2, pfc_b2, pfb);
  cls1_k<<<B_, 512, 0, stream>>>(maxf, meanf, pfb, cls_w1, cls_b1, c1);
  cls2_k<<<B_, 256, 0, stream>>>(c1, cls_w2, cls_b2, c2);
  cls3_k<<<B_, 32, 0, stream>>>(c2, cls_w3, cls_b3, (float*)d_out);
}
